// Luong_Concat_Attention_6545530159368
// MI455X (gfx1250) — matrix operand order repaired and checked by reading, not compiled, not run on hardware
//
#include <hip/hip_runtime.h>
#include <hip/hip_bf16.h>

#define MODEL_D 512
#define ATT_A   512
#define BATCH   32
#define SEQ     4096

typedef __attribute__((ext_vector_type(16))) __bf16 v16bf;
typedef __attribute__((ext_vector_type(8)))  __bf16 v8bf;
typedef __attribute__((ext_vector_type(4)))  __bf16 v4bf;
typedef __attribute__((ext_vector_type(8)))  float  v8f;
typedef __attribute__((ext_vector_type(4)))  float  v4f;
typedef unsigned int u32x4 __attribute__((ext_vector_type(4)));
typedef int i32x8 __attribute__((ext_vector_type(8)));
typedef int i32x4 __attribute__((ext_vector_type(4)));

// ---------------------------------------------------------------------------
// TDM issue: DMA a contiguous tile of n8*8 bytes from global into LDS.
// Descriptor per CDNA5 ISA (1 valid D#,
// data_size=8B, 1 row). Caller is responsible for s_wait_tensorcnt.
// TDM ignores EXEC; call from one wave only.
// ---------------------------------------------------------------------------
__device__ inline void tdm_issue(const void* gsrc, void* ldst, int n8) {
#if __has_builtin(__builtin_amdgcn_tensor_load_to_lds)
  unsigned long long ga = (unsigned long long)(size_t)gsrc;
  unsigned int la = (unsigned int)(size_t)ldst;   // low 32 bits = LDS offset
  u32x4 g0;
  g0[0] = 1u;                                     // count=1, user mode
  g0[1] = la;                                     // lds_addr
  g0[2] = (unsigned int)ga;                       // global_addr[31:0]
  g0[3] = (unsigned int)((ga >> 32) & 0x1FFFFFFull) | (2u << 30); // hi | type=2
  i32x8 g1;
  g1[0] = 3 << 16;                                // data_size = 8B
  g1[1] = (n8 & 0xFFFF) << 16;                    // tensor_dim0 lo16
  g1[2] = ((n8 >> 16) & 0xFFFF) | (1 << 16);      // tensor_dim0 hi | tensor_dim1=1
  g1[3] = (n8 & 0xFFFF) << 16;                    // tile_dim0
  g1[4] = 1;                                      // tile_dim1 = 1
  g1[5] = n8;                                     // tensor_dim0_stride
  g1[6] = 0;
  g1[7] = 0;
  i32x4 gz = {0, 0, 0, 0};
#if __clang_major__ >= 23
  i32x8 gz8 = {0, 0, 0, 0, 0, 0, 0, 0};
  __builtin_amdgcn_tensor_load_to_lds(g0, g1, gz, gz, gz8, 0);
#else
  __builtin_amdgcn_tensor_load_to_lds(g0, g1, gz, gz, 0);
#endif
#else
  // Fallback: synchronous vector copy by one wave (32 lanes x 16B)
  int lane = threadIdx.x & 31;
  const v4f* src = (const v4f*)gsrc;
  v4f* dst = (v4f*)ldst;
  for (int i = lane; i < n8 / 2; i += 32) dst[i] = src[i];
#endif
}

__device__ inline void tdm_wait_le1() {
#if __has_builtin(__builtin_amdgcn_s_wait_tensorcnt)
  __builtin_amdgcn_s_wait_tensorcnt(1);
#endif
}
__device__ inline void tdm_wait_0() {
#if __has_builtin(__builtin_amdgcn_s_wait_tensorcnt)
  __builtin_amdgcn_s_wait_tensorcnt(0);
#endif
}

// ---------------------------------------------------------------------------
// Kernel 1: W_e (d,a) f32 -> W_t (a,d) bf16 (N-major so B-fragments are
// contiguous K-runs, matching the WMMA B-matrix VGPR layout).
// ---------------------------------------------------------------------------
__global__ void transpose_we_kernel(const float* __restrict__ W,
                                    __bf16* __restrict__ Wt) {
  int idx = blockIdx.x * blockDim.x + threadIdx.x;
  int stride = gridDim.x * blockDim.x;
  for (int i = idx; i < MODEL_D * ATT_A; i += stride) {
    int a = i >> 9;
    int d = i & 511;
    Wt[(size_t)a * MODEL_D + d] = (__bf16)W[(size_t)d * ATT_A + a];
  }
}

// ---------------------------------------------------------------------------
// Kernel 2: dec_proj[b,a] = decode_state[b,:] . W_d[:,a]   (tiny GEMV)
// ---------------------------------------------------------------------------
__global__ void dec_proj_kernel(const float* __restrict__ dec,
                                const float* __restrict__ W,
                                float* __restrict__ decp) {
  __shared__ float sdec[MODEL_D];
  int b = blockIdx.x, tid = threadIdx.x;
  for (int i = tid; i < MODEL_D; i += blockDim.x) sdec[i] = dec[b * MODEL_D + i];
  __syncthreads();
  for (int a = tid; a < ATT_A; a += blockDim.x) {
    float acc = 0.f;
    #pragma unroll 8
    for (int d = 0; d < MODEL_D; ++d)
      acc = fmaf(sdec[d], W[(size_t)(MODEL_D + d) * ATT_A + a], acc);
    decp[b * ATT_A + a] = acc;
  }
}

// ---------------------------------------------------------------------------
// Kernel 3 (main): fused enc_proj GEMM (bf16 WMMA) + tanh + v-dot -> scores.
// 128 threads = 4 waves. WG owns 64 S-rows; wave w owns rows w*16..w*16+15.
// A slab (64x512 bf16, 64KB) in LDS; W_t streamed in 32-column chunks (32KB)
// via double-buffered TDM so DMA overlaps the 32-WMMA compute phase.
// Each A-fragment load feeds TWO WMMAs (two N-tiles) -> 1.5KB LDS per WMMA.
// ---------------------------------------------------------------------------
__global__ __launch_bounds__(128) void attn_scores_kernel(
    const float* __restrict__ enc, const __bf16* __restrict__ Wt,
    const float* __restrict__ decp, const float* __restrict__ vvec,
    float* __restrict__ scores) {
  extern __shared__ char smem[];
  __bf16* ldsA   = (__bf16*)smem;                                   // 64KB
  char*   ldsB   = smem + 64 * MODEL_D * 2;                         // 2x32KB
  float*  ldsDec = (float*)(smem + 64 * MODEL_D * 2 + 64 * MODEL_D * 2);
  float*  ldsV   = ldsDec + ATT_A;

  const int b = blockIdx.y;
  const int sbase = blockIdx.x * 64;
  const int tid = threadIdx.x;
  const int lane = tid & 31;
  const int wave = tid >> 5;
  const int hi = lane >> 4;     // which 16-lane half
  const int lg = lane & 15;

  for (int i = tid; i < ATT_A; i += 128) {
    ldsDec[i] = decp[b * ATT_A + i];
    ldsV[i]   = vvec[i];
  }

  // Load 64x512 f32 slab of encode_state, convert to bf16 into LDS.
  const float* encSlab = enc + ((size_t)b * SEQ + sbase) * MODEL_D;
  for (int i = tid * 4; i < 64 * MODEL_D; i += 128 * 4) {
    v4f x = *(const v4f*)(encSlab + i);
    v4bf y;
    y[0] = (__bf16)x[0]; y[1] = (__bf16)x[1];
    y[2] = (__bf16)x[2]; y[3] = (__bf16)x[3];
    *(v4bf*)(ldsA + i) = y;
  }

  // Prime the B pipeline: chunk 0 (32 columns of W_t = 32KB) into buffer 0.
  if (wave == 0) tdm_issue(Wt, ldsB, 4096);
  __syncthreads();

  float sacc[8] = {0.f, 0.f, 0.f, 0.f, 0.f, 0.f, 0.f, 0.f};

  // A fragment base: lane holds row (wave*16 + lg); halves split K by 16.
  const char* aBase = (const char*)ldsA + (size_t)(wave * 16 + lg) * (MODEL_D * 2) + hi * 32;

  const int NIT = ATT_A / 32;   // 16 iterations x 32 columns
  for (int it = 0; it < NIT; ++it) {
    char* cur = ldsB + (size_t)(it & 1) * (32 * MODEL_D * 2);
    char* nxt = ldsB + (size_t)((it + 1) & 1) * (32 * MODEL_D * 2);
    if (wave == 0) {
      if (it + 1 < NIT) {
        tdm_issue(Wt + (size_t)(it + 1) * 32 * MODEL_D, nxt, 4096);
        tdm_wait_le1();   // in-order TDM: <=1 outstanding => cur complete
      } else {
        tdm_wait_0();
      }
    }
    __syncthreads();            // cur visible to all waves

    // B fragment bases: lane holds column lg / lg+16; halves split K by 16.
    const char* bBase0 = cur + (size_t)lg * (MODEL_D * 2) + hi * 32;
    const char* bBase1 = bBase0 + 16 * MODEL_D * 2;

    v8f acc0 = {0.f, 0.f, 0.f, 0.f, 0.f, 0.f, 0.f, 0.f};
    v8f acc1 = {0.f, 0.f, 0.f, 0.f, 0.f, 0.f, 0.f, 0.f};
    #pragma unroll
    for (int k = 0; k < MODEL_D; k += 32) {
      union { v16bf v; v8bf h[2]; } af, b0, b1;
      af.h[0] = *(const v8bf*)(aBase + k * 2);
      af.h[1] = *(const v8bf*)(aBase + k * 2 + 16);
      b0.h[0] = *(const v8bf*)(bBase0 + k * 2);
      b0.h[1] = *(const v8bf*)(bBase0 + k * 2 + 16);
      b1.h[0] = *(const v8bf*)(bBase1 + k * 2);
      b1.h[1] = *(const v8bf*)(bBase1 + k * 2 + 16);
      acc0 = __builtin_amdgcn_wmma_f32_16x16x32_bf16(
          false, af.v, false, b0.v, (short)0, acc0, false, false);
      acc1 = __builtin_amdgcn_wmma_f32_16x16x32_bf16(
          false, af.v, false, b1.v, (short)0, acc1, false, false);
    }

    // Epilogue: accX[j] = proj[row=wave*16+j+8*hi][col], col0=n+lg, col1=n+16+lg
    const int n = it * 32;
    const float dp0 = ldsDec[n + lg];
    const float vw0 = ldsV[n + lg];
    const float dp1 = ldsDec[n + 16 + lg];
    const float vw1 = ldsV[n + 16 + lg];
    #pragma unroll
    for (int j = 0; j < 8; ++j) {
      sacc[j] += tanhf(acc0[j] + dp0) * vw0;
      sacc[j] += tanhf(acc1[j] + dp1) * vw1;
    }
    __syncthreads();            // all reads of cur done before it is re-filled
  }

  // Reduce over the 16 lanes of each half (columns) -> per-row scores.
  #pragma unroll
  for (int j = 0; j < 8; ++j) {
    float val = sacc[j];
    val += __shfl_xor(val, 1, 32);
    val += __shfl_xor(val, 2, 32);
    val += __shfl_xor(val, 4, 32);
    val += __shfl_xor(val, 8, 32);
    if (lg == 0) {
      int row = wave * 16 + j + hi * 8;
      scores[(size_t)b * SEQ + sbase + row] = val;
    }
  }
}

// ---------------------------------------------------------------------------
// Kernel 4: per-batch softmax stats (max, sum-of-exp).
// ---------------------------------------------------------------------------
__global__ void softmax_stats_kernel(const float* __restrict__ scores,
                                     float* __restrict__ stats) {
  __shared__ float red[256];
  int b = blockIdx.x, tid = threadIdx.x;
  const float* row = scores + (size_t)b * SEQ;
  float m = -3.4e38f;
  for (int i = tid; i < SEQ; i += 256) m = fmaxf(m, row[i]);
  red[tid] = m; __syncthreads();
  for (int s = 128; s > 0; s >>= 1) {
    if (tid < s) red[tid] = fmaxf(red[tid], red[tid + s]);
    __syncthreads();
  }
  m = red[0]; __syncthreads();
  float z = 0.f;
  for (int i = tid; i < SEQ; i += 256) z += __expf(row[i] - m);
  red[tid] = z; __syncthreads();
  for (int s = 128; s > 0; s >>= 1) {
    if (tid < s) red[tid] += red[tid + s];
    __syncthreads();
  }
  if (tid == 0) { stats[b * 2] = m; stats[b * 2 + 1] = red[0]; }
}

// ---------------------------------------------------------------------------
// Kernel 5: context[b,d] = sum_s softmax(scores)[b,s] * enc[b,s,d]
// Memory-bound streaming pass (268MB @ 23.3TB/s ~ 12us); prefetch ahead.
// ---------------------------------------------------------------------------
__global__ __launch_bounds__(512) void context_kernel(
    const float* __restrict__ enc, const float* __restrict__ scores,
    const float* __restrict__ stats, float* __restrict__ ctx) {
  __shared__ float pbuf[256];
  int b = blockIdx.x, d = threadIdx.x;
  float m = stats[b * 2];
  float invZ = 1.0f / stats[b * 2 + 1];
  const float* encB = enc + (size_t)b * SEQ * MODEL_D;
  const float* srow = scores + (size_t)b * SEQ;
  float acc = 0.f;
  for (int s0 = 0; s0 < SEQ; s0 += 256) {
    __syncthreads();
    if (d < 256) pbuf[d] = __expf(srow[s0 + d] - m) * invZ;
    __syncthreads();
    #pragma unroll 4
    for (int j = 0; j < 256; ++j) {
      if ((j & 31) == 0 && (s0 + j + 32) < SEQ)
        __builtin_prefetch(encB + (size_t)(s0 + j + 32) * MODEL_D + d, 0, 0);
      acc += pbuf[j] * encB[(size_t)(s0 + j) * MODEL_D + d];
    }
  }
  ctx[(size_t)b * MODEL_D + d] = acc;
}

// ---------------------------------------------------------------------------
extern "C" void kernel_launch(void* const* d_in, const int* in_sizes, int n_in,
                              void* d_out, int out_size, void* d_ws, size_t ws_size,
                              hipStream_t stream) {
  (void)in_sizes; (void)n_in; (void)out_size; (void)ws_size;
  const float* enc = (const float*)d_in[0];   // (32, 4096, 512)
  const float* dec = (const float*)d_in[1];   // (32, 512)
  const float* W   = (const float*)d_in[2];   // (1024, 512)
  const float* v   = (const float*)d_in[3];   // (512,)
  float* out = (float*)d_out;
  float* scores = out;                               // 32*4096
  float* ctx = out + (size_t)BATCH * SEQ;            // 32*512

  char* ws = (char*)d_ws;
  __bf16* Wt   = (__bf16*)ws;                                            // 512KB
  float*  decp = (float*)(ws + (size_t)MODEL_D * ATT_A * 2);             // 64KB
  float*  stats = (float*)(ws + (size_t)MODEL_D * ATT_A * 2
                              + (size_t)BATCH * ATT_A * 4);              // 256B

  transpose_we_kernel<<<dim3(256), dim3(256), 0, stream>>>(W, Wt);
  dec_proj_kernel<<<dim3(BATCH), dim3(256), 0, stream>>>(dec, W, decp);

  size_t lds_bytes = (size_t)64 * MODEL_D * 2   // A slab          (64KB)
                   + (size_t)64 * MODEL_D * 2   // B double buffer (64KB)
                   + (size_t)2 * ATT_A * 4;     // dec row + v     (4KB)
  attn_scores_kernel<<<dim3(SEQ / 64, BATCH), dim3(128), lds_bytes, stream>>>(
      enc, Wt, decp, v, scores);

  softmax_stats_kernel<<<dim3(BATCH), dim3(256), 0, stream>>>(scores, stats);
  context_kernel<<<dim3(BATCH), dim3(512), 0, stream>>>(enc, scores, stats, ctx);
}
